// C_SCNN_17918603559439
// MI455X (gfx1250) — compile-verified, matrix operand order checked by reading, not run on hardware
//
#include <hip/hip_runtime.h>
#include <hip/hip_bf16.h>
#include <stdint.h>

// Problem dims
#define BB 8
#define CC 256
#define HH 25
#define WW 512

typedef __bf16 bf16x16 __attribute__((ext_vector_type(16)));
typedef float  f32x8   __attribute__((ext_vector_type(8)));

union FragU { uint4 u[2]; bf16x16 v; };

__device__ __forceinline__ unsigned short f2bf(float x) {
  unsigned u = __float_as_uint(x);
  unsigned r = u + 0x7FFFu + ((u >> 16) & 1u);   // round-to-nearest-even
  return (unsigned short)(r >> 16);
}

__device__ __forceinline__ f32x8 wmma_bf16(bf16x16 a, bf16x16 b, f32x8 c) {
  // (neg_a, A, neg_b, B, c_mod, C, reuse_a, reuse_b)
  return __builtin_amdgcn_wmma_f32_16x16x32_bf16(false, a, false, b, (short)0, c, false, false);
}

// A-fragment (16x32 bf16, row-major, this lane's row precomputed by caller).
// Lane L: row = L%16, kbase = (L<16)?0:8 ; v0..3 <- K kbase..kbase+7, v4..7 <- K 16+kbase..
__device__ __forceinline__ bf16x16 load_frag_lds(const unsigned short* rowBase, int kt, int lane) {
  int kb = (lane & 16) ? 8 : 0;
  const unsigned short* p = rowBase + kt * 32 + kb;
  FragU f;
  f.u[0] = *(const uint4*)(p);        // ds_load_b128
  f.u[1] = *(const uint4*)(p + 16);   // ds_load_b128
  return f.v;
}

// B-fragment from pre-swizzled global weights: 8 contiguous dwords per lane.
__device__ __forceinline__ bf16x16 load_frag_Bswz(const unsigned int* swz, int kt, int nt, int lane) {
  const unsigned int* p = swz + ((((kt << 4) + nt) * 32 + lane) << 3);
  FragU f;
  f.u[0] = *(const uint4*)(p);        // global_load_b128 (L2-resident)
  f.u[1] = *(const uint4*)(p + 4);
  return f.v;
}

// ---------------------------------------------------------------------------
// Weight swizzle: tap 0..8 = w_up2[:,:,ky,kx] ; tap 9 = w_msg[:,:,4,0].
// B[k][n] = Wmat[n][k]; dword (kt,nt,lane,j) packs B[k],B[k+1] (bf16 lo/hi).
// ---------------------------------------------------------------------------
__global__ void prep_kernel(const float* __restrict__ wmsg, const float* __restrict__ wup2,
                            unsigned int* __restrict__ Mswz, unsigned int* __restrict__ Wswz)
{
  int idx = blockIdx.x * 256 + threadIdx.x;     // 10 * 32768 total
  int tap = idx >> 15;
  int r   = idx & 32767;
  int j    = r & 7;
  int lane = (r >> 3) & 31;
  int nt   = (r >> 8) & 15;
  int kt   = r >> 12;
  int n  = nt * 16 + (lane & 15);
  int kb = (lane & 16) ? 8 : 0;
  int k  = kt * 32 + ((j < 4) ? (kb + 2 * j) : (16 + kb + 2 * (j - 4)));
  float lo, hi;
  if (tap == 9) {
    lo = wmsg[(n * 256 + k) * 9 + 4];
    hi = wmsg[(n * 256 + k + 1) * 9 + 4];
  } else {
    int ky = tap / 3, kx = tap - ky * 3;
    lo = wup2[((n * 256 + k) * 3 + ky) * 3 + kx];
    hi = wup2[((n * 256 + k + 1) * 3 + ky) * 3 + kx];
  }
  unsigned int d = (unsigned)f2bf(lo) | ((unsigned)f2bf(hi) << 16);
  if (tap == 9) Mswz[r] = d;
  else          Wswz[tap * 32768 + r] = d;
}

// Zero BN stat accumulators; write identity affine (scale=1, shift=0).
__global__ void init_misc(float* stats1, float* stats2, float* scid, float* shid) {
  int t = threadIdx.x;
  stats1[t] = 0.f; stats2[t] = 0.f;
  if (t < 256) { scid[t] = 1.0f; shid[t] = 0.0f; }
}

// ---------------------------------------------------------------------------
// SCNN pass: out[h'] with h' = H-1-h (the reference reverses H), where
// cur = in_row(h) + relu(M x prev).  prev initialized to 0 makes h=0 exact.
// Only `prev` lives in LDS (ping-pong, ONE barrier/step); each lane loads its
// own xi slice (2 channels x 8 consecutive W) as 128-bit loads, with the BN
// affine hoisted per-lane.  Next row's loads issue at loop bottom so they
// overlap the following step's WMMA chain.  A-fragments double-buffered.
// WG: 16 (b,w) rows x 256 channels; 8 waves x 2 N-tiles; K=256 in 8 WMMAs.
// ---------------------------------------------------------------------------
__global__ void __launch_bounds__(256) scnn_kernel(
    const float* __restrict__ xin, float* __restrict__ xout,
    const unsigned int* __restrict__ Mswz,
    const float* __restrict__ sc, const float* __restrict__ sh,
    int relu_in, float* __restrict__ stats)
{
  __shared__ __align__(16) unsigned short prevA[16 * 264];
  __shared__ __align__(16) unsigned short prevB[16 * 264];

  const int tid  = threadIdx.x;
  const int b    = blockIdx.x >> 5;          // 32 w-tiles per batch
  const int w0   = (blockIdx.x & 31) << 4;
  const int wave = tid >> 5;
  const int lane = tid & 31;
  const int half8 = (lane & 16) ? 8 : 0;
  const int nt0 = wave * 2, nt1 = wave * 2 + 1;
  const int ch0 = nt0 * 16 + (lane & 15);
  const int ch1 = nt1 * 16 + (lane & 15);

  for (int i = tid; i < 16 * 264; i += 256) prevA[i] = 0;

  // loop-invariant per-lane BN affine for this lane's two channels
  const float sA = sc[ch0], bA = sh[ch0];
  const float sB = sc[ch1], bB = sh[ch1];

  const float* xi0p = xin  + ((size_t)(b * CC + ch0) * HH) * WW + w0 + half8;
  const float* xi1p = xin  + ((size_t)(b * CC + ch1) * HH) * WW + w0 + half8;
  float*       xo0p = xout + ((size_t)(b * CC + ch0) * HH) * WW + w0 + half8;
  float*       xo1p = xout + ((size_t)(b * CC + ch1) * HH) * WW + w0 + half8;

  // prologue: row 0 xi (four global_load_b128 per lane)
  float4 x0a = *(const float4*)(xi0p);
  float4 x0b = *(const float4*)(xi0p + 4);
  float4 x1a = *(const float4*)(xi1p);
  float4 x1b = *(const float4*)(xi1p + 4);

  float s0 = 0.f, q0 = 0.f, s1 = 0.f, q1 = 0.f;

  for (int h = 0; h < HH; ++h) {
    unsigned short* prevCur = (h & 1) ? prevB : prevA;
    unsigned short* prevNxt = (h & 1) ? prevA : prevB;
    __syncthreads();   // prevCur complete; prevNxt free to write below

    f32x8 acc0 = {}; f32x8 acc1 = {};
    const unsigned short* arow = &prevCur[(lane & 15) * 264];
    bf16x16 a = load_frag_lds(arow, 0, lane);
    #pragma unroll
    for (int kt = 0; kt < 8; ++kt) {
      bf16x16 an = (kt < 7) ? load_frag_lds(arow, kt + 1, lane) : a;  // dbl-buffer
      acc0 = wmma_bf16(a, load_frag_Bswz(Mswz, kt, nt0, lane), acc0);
      acc1 = wmma_bf16(a, load_frag_Bswz(Mswz, kt, nt1, lane), acc1);
      a = an;
    }

    const float xi0[8] = {x0a.x, x0a.y, x0a.z, x0a.w, x0b.x, x0b.y, x0b.z, x0b.w};
    const float xi1[8] = {x1a.x, x1a.y, x1a.z, x1a.w, x1b.x, x1b.y, x1b.z, x1b.w};
    float r0[8], r1[8];
    #pragma unroll
    for (int j = 0; j < 8; ++j) {
      float a0 = sA * xi0[j] + bA;
      float a1 = sB * xi1[j] + bB;
      if (relu_in) { a0 = fmaxf(a0, 0.f); a1 = fmaxf(a1, 0.f); }
      float v0 = a0 + fmaxf(acc0[j], 0.f);   // xi stays full fp32
      float v1 = a1 + fmaxf(acc1[j], 0.f);
      r0[j] = v0; r1[j] = v1;
      s0 += v0; q0 += v0 * v0;
      s1 += v1; q1 += v1 * v1;
      prevNxt[(j + half8) * 264 + ch0] = f2bf(v0);   // other buffer: safe
      prevNxt[(j + half8) * 264 + ch1] = f2bf(v1);
    }
    // packed 128-bit global stores (each lane's 8 results are W-consecutive)
    {
      float* p0 = xo0p + (size_t)(HH - 1 - h) * WW;
      float* p1 = xo1p + (size_t)(HH - 1 - h) * WW;
      *(float4*)(p0)     = make_float4(r0[0], r0[1], r0[2], r0[3]);
      *(float4*)(p0 + 4) = make_float4(r0[4], r0[5], r0[6], r0[7]);
      *(float4*)(p1)     = make_float4(r1[0], r1[1], r1[2], r1[3]);
      *(float4*)(p1 + 4) = make_float4(r1[4], r1[5], r1[6], r1[7]);
    }
    // issue next row's xi loads now; they overlap the next WMMA chain
    if (h + 1 < HH) {
      const float* n0 = xi0p + (size_t)(h + 1) * WW;
      const float* n1 = xi1p + (size_t)(h + 1) * WW;
      x0a = *(const float4*)(n0); x0b = *(const float4*)(n0 + 4);
      x1a = *(const float4*)(n1); x1b = *(const float4*)(n1 + 4);
      if (h + 2 < HH) {
        __builtin_prefetch(xi0p + (size_t)(h + 2) * WW, 0, 0);  // global_prefetch_b8
        __builtin_prefetch(xi1p + (size_t)(h + 2) * WW, 0, 0);
      }
    }
  }

  atomicAdd(&stats[ch0],       s0);
  atomicAdd(&stats[256 + ch0], q0);
  atomicAdd(&stats[ch1],       s1);
  atomicAdd(&stats[256 + ch1], q1);
}

__global__ void bnscale_kernel(const float* __restrict__ stats,
                               const float* __restrict__ gamma, const float* __restrict__ beta,
                               float* __restrict__ sc, float* __restrict__ sh)
{
  int c = threadIdx.x;
  const float n = (float)(BB * HH * WW);
  float mean = stats[c] / n;
  float var  = stats[256 + c] / n - mean * mean;
  float s = gamma[c] * rsqrtf(var + 1e-5f);
  sc[c] = s;
  sh[c] = beta[c] - mean * s;
}

// ---------------------------------------------------------------------------
// 3x3 conv as implicit GEMM over a staged halo tile of
// u = up_w(relu(bn2(x))) at width 1024.  Epilogue folds the w_conv1 channel
// reduction (linear ops commute with up_w), producing z[b][h][w2] (819 KB).
// ---------------------------------------------------------------------------
__global__ void __launch_bounds__(256) conv_kernel(
    const float* __restrict__ xin,          // scnn2 raw output
    const unsigned int* __restrict__ Wswz,  // 9 swizzled taps
    const float* __restrict__ sc, const float* __restrict__ sh,
    const float* __restrict__ v1,           // w_conv1 (256)
    float* __restrict__ z)                  // (B,25,1024)
{
  __shared__ __align__(16) unsigned short uT[3 * 18 * 264];
  __shared__ float zred[16];

  const int tid = threadIdx.x;
  const int wt  = blockIdx.x & 63;
  const int h   = (blockIdx.x >> 6) % 25;
  const int b   = blockIdx.x / (64 * 25);
  const int w20 = wt << 4;

  if (tid < 16) zred[tid] = 0.f;

  // stage u halo tile [hh 0..2][wi 0..17][c 0..255], zero-padded edges;
  // BN+ReLU applied to source samples BEFORE interpolation (matches ref).
  for (int e = tid; e < 3 * 18 * 256; e += 256) {
    int wi   = e % 18;
    int rest = e / 18;
    int c    = rest & 255;
    int hh   = rest >> 8;
    int hs = h + hh - 1;
    int w2 = w20 + wi - 1;
    float v = 0.f;
    if (hs >= 0 && hs < HH && w2 >= 0 && w2 < 1024) {
      int   num = w2 * 511;
      int   i0  = num / 1023;
      float f   = (float)(num - i0 * 1023) * (1.0f / 1023.0f);
      int   i1  = min(i0 + 1, 511);
      const float* base = xin + ((b * CC + c) * HH + hs) * WW;
      float scc = sc[c], shc = sh[c];
      float x0 = fmaxf(scc * base[i0] + shc, 0.f);
      float x1 = fmaxf(scc * base[i1] + shc, 0.f);
      v = x0 + f * (x1 - x0);
    }
    uT[(hh * 18 + wi) * 264 + c] = f2bf(v);
  }
  __syncthreads();

  const int wave = tid >> 5;
  const int lane = tid & 31;
  const int nt0 = wave * 2, nt1 = nt0 + 1;
  f32x8 acc0 = {}; f32x8 acc1 = {};

  #pragma unroll
  for (int ky = 0; ky < 3; ++ky) {
    #pragma unroll
    for (int kx = 0; kx < 3; ++kx) {
      const unsigned int* bsw = Wswz + (ky * 3 + kx) * 32768;
      const unsigned short* arow = &uT[(ky * 18 + (lane & 15) + kx) * 264];
      bf16x16 a = load_frag_lds(arow, 0, lane);
      #pragma unroll
      for (int kt = 0; kt < 8; ++kt) {
        bf16x16 an = (kt < 7) ? load_frag_lds(arow, kt + 1, lane) : a;
        acc0 = wmma_bf16(a, load_frag_Bswz(bsw, kt, nt0, lane), acc0);
        acc1 = wmma_bf16(a, load_frag_Bswz(bsw, kt, nt1, lane), acc1);
        a = an;
      }
    }
  }

  const int half8 = (lane & 16) ? 8 : 0;
  const int ch0 = nt0 * 16 + (lane & 15);
  const int ch1 = nt1 * 16 + (lane & 15);
  float g0 = v1[ch0], g1 = v1[ch1];
  #pragma unroll
  for (int j = 0; j < 8; ++j) {
    atomicAdd(&zred[j + half8], acc0[j] * g0 + acc1[j] * g1);  // ds_add_f32
  }
  __syncthreads();
  if (tid < 16) z[(b * 25 + h) * 1024 + w20 + tid] = zred[tid];
}

// ---------------------------------------------------------------------------
// Tail: out[b][w4] = sigmoid( sum_h v2[h] * up(up(z))[b][h][w4] )
// (two explicit lerp levels, exact integer index math)
// ---------------------------------------------------------------------------
__global__ void final_kernel(const float* __restrict__ z, const float* __restrict__ v2,
                             float* __restrict__ out)
{
  int idx = blockIdx.x * 256 + threadIdx.x;
  if (idx >= BB * 4096) return;
  int b  = idx >> 12;
  int w4 = idx & 4095;

  int   num4 = w4 * 2047;
  int   i40  = num4 / 4095;
  float f4   = (float)(num4 - i40 * 4095) * (1.0f / 4095.0f);
  int   i41  = min(i40 + 1, 2047);

  int n3a = i40 * 1023; int ja0 = n3a / 2047;
  float fa = (float)(n3a - ja0 * 2047) * (1.0f / 2047.0f);
  int ja1 = min(ja0 + 1, 1023);
  int n3b = i41 * 1023; int jb0 = n3b / 2047;
  float fb = (float)(n3b - jb0 * 2047) * (1.0f / 2047.0f);
  int jb1 = min(jb0 + 1, 1023);

  float acc = 0.f;
  #pragma unroll 5
  for (int h = 0; h < HH; ++h) {
    const float* zh = z + (b * 25 + h) * 1024;
    float za = zh[ja0] + fa * (zh[ja1] - zh[ja0]);
    float zb = zh[jb0] + fb * (zh[jb1] - zh[jb0]);
    acc += v2[h] * (za + f4 * (zb - za));
  }
  out[idx] = 1.0f / (1.0f + __expf(-acc));
}

// ---------------------------------------------------------------------------
extern "C" void kernel_launch(void* const* d_in, const int* in_sizes, int n_in,
                              void* d_out, int out_size, void* d_ws, size_t ws_size,
                              hipStream_t stream)
{
  const float* p2c   = (const float*)d_in[0];
  const float* wmsg  = (const float*)d_in[1];
  const float* gamma = (const float*)d_in[2];
  const float* beta  = (const float*)d_in[3];
  const float* wup2  = (const float*)d_in[4];
  const float* v1    = (const float*)d_in[5];
  const float* v2    = (const float*)d_in[6];

  char* ws = (char*)d_ws;
  size_t off = 0;
  auto walloc = [&](size_t bytes) -> void* {
    void* p = ws + off;
    off = (off + bytes + 255) & ~(size_t)255;
    return p;
  };
  unsigned int* Mswz = (unsigned int*)walloc(32768u * 4);
  unsigned int* Wswz = (unsigned int*)walloc(9u * 32768u * 4);
  float* stats1 = (float*)walloc(512 * 4);
  float* stats2 = (float*)walloc(512 * 4);
  float* sc1  = (float*)walloc(256 * 4);
  float* sh1  = (float*)walloc(256 * 4);
  float* sc2  = (float*)walloc(256 * 4);
  float* sh2  = (float*)walloc(256 * 4);
  float* scid = (float*)walloc(256 * 4);
  float* shid = (float*)walloc(256 * 4);
  float* z1024 = (float*)walloc((size_t)BB * 25 * 1024 * 4);
  float* out1  = (float*)walloc((size_t)BB * CC * HH * WW * 4);
  float* out2  = (float*)walloc((size_t)BB * CC * HH * WW * 4);

  prep_kernel<<<1280, 256, 0, stream>>>(wmsg, wup2, Mswz, Wswz);
  init_misc<<<1, 512, 0, stream>>>(stats1, stats2, scid, shid);

  // pass 1: raw input (identity affine, no relu on input)
  scnn_kernel<<<256, 256, 0, stream>>>(p2c, out1, Mswz, scid, shid, 0, stats1);
  bnscale_kernel<<<1, 256, 0, stream>>>(stats1, gamma, beta, sc1, sh1);

  // pass 2: input = relu(bn1(out1))
  scnn_kernel<<<256, 256, 0, stream>>>(out1, out2, Mswz, sc1, sh1, 1, stats2);
  bnscale_kernel<<<1, 256, 0, stream>>>(stats2, gamma, beta, sc2, sh2);

  // conv (+fused up_w in, + channel reduction out)
  conv_kernel<<<BB * 25 * 64, 256, 0, stream>>>(out2, Wswz, sc2, sh2, v1, z1024);

  // tail: double-upsample + H reduction + sigmoid
  final_kernel<<<128, 256, 0, stream>>>(z1024, v2, (float*)d_out);
}
